// GAT_59742995087950
// MI455X (gfx1250) — matrix operand order repaired and checked by reading, not compiled, not run on hardware
//
#include <hip/hip_runtime.h>
#include <hip/hip_bf16.h>

// ---------------------------------------------------------------------------
// GATv2 (2 layers) for MI455X / gfx1250, wave32.
//   N=50000, E=800000 (+N self loops), DIM_IN=128, H=8, C1=16, DIM_OUT=64.
// GEMMs via v_wmma_f32_16x16x32_f16 with f16 operands pre-packed so every
// A/B fragment is a contiguous global_load_b128 pair (no scalar DS traffic,
// no f32->f16 conversion in the hot loop).
// Edge phase: wave-per-edge, coalesced float4 row gathers (L2-resident),
// shfl_xor reductions, global_atomic_add_f32 segment sums.
// Segment-max is skipped: softmax is shift-invariant and logits are O(1).
// ---------------------------------------------------------------------------

typedef __attribute__((ext_vector_type(16))) _Float16 v16h;
typedef __attribute__((ext_vector_type(8)))  _Float16 v8h;
typedef __attribute__((ext_vector_type(4)))  _Float16 v4h;
typedef __attribute__((ext_vector_type(8)))  float    v8f;

__device__ __forceinline__ float lrelu(float v) { return v > 0.f ? v : 0.2f * v; }

__device__ __forceinline__ void atomAdd(float* p, float v) {
    __hip_atomic_fetch_add(p, v, __ATOMIC_RELAXED, __HIP_MEMORY_SCOPE_AGENT);
}

// ---------------------------------------------------------------------------
// One-time packing kernels (tiny).
// ---------------------------------------------------------------------------

// f32 row-major [n,128] -> f16 row-major, 4 elements per thread.
__global__ __launch_bounds__(256)
void f32_to_f16_4(const float* __restrict__ in, _Float16* __restrict__ out, int total4) {
    const int i = blockIdx.x * 256 + threadIdx.x;
    if (i >= total4) return;
    const float4 v = ((const float4*)in)[i];
    v4h o;
    o[0] = (_Float16)v.x; o[1] = (_Float16)v.y;
    o[2] = (_Float16)v.z; o[3] = (_Float16)v.w;
    ((v4h*)out)[i] = o;
}

// W [K,C] f32 row-major -> WT [C,K] f16 (k contiguous per column).
__global__ __launch_bounds__(256)
void pack_w_t(const float* __restrict__ W, _Float16* __restrict__ WT, int K, int C) {
    const int i = blockIdx.x * 256 + threadIdx.x;   // i = col*K + k
    if (i >= K * C) return;
    const int col = i / K, k = i - col * K;
    WT[i] = (_Float16)W[(long)k * C + col];
}

// ---------------------------------------------------------------------------
// Dual GEMM: Ya = X @ Wa, Yb = X @ Wb.  X16:[n,128] f16 row-major,
// WaT/WbT:[C,128] f16 column-major (k contiguous).  K fixed at 128.
// Block = 128 threads (4 waves); wave w owns row-tile blockIdx.x*4+w,
// col-tile blockIdx.y (16 cols).  All fragments are contiguous b128 loads.
// ---------------------------------------------------------------------------
__global__ __launch_bounds__(128)
void gemm_dual_wmma(const _Float16* __restrict__ X16,
                    const _Float16* __restrict__ WaT,
                    const _Float16* __restrict__ WbT,
                    float* __restrict__ Ya,
                    float* __restrict__ Yb,
                    int nRowTiles, int C) {
    constexpr int K = 128;
    const int t = threadIdx.x;
    const int wave = t >> 5, lane = t & 31;
    const int rowTile = blockIdx.x * 4 + wave;
    if (rowTile >= nRowTiles) return;          // whole-wave uniform exit

    const int m   = lane & 15;                 // row within A-tile / col within B
    const int grp = lane >> 4;                 // lane group 0/1
    const int col = blockIdx.y * 16 + m;

    const _Float16* xrow  = X16 + (long)(rowTile * 16 + m) * K;
    const _Float16* wacol = WaT + (long)col * K;
    const _Float16* wbcol = WbT + (long)col * K;

    v8f ca = {}; v8f cb = {};
#pragma unroll
    for (int ks = 0; ks < 4; ++ks) {
        const int k0 = ks * 32;
        // A fragment (contiguous k order, same as B): grp0 K = k0+[0..15];
        // grp1 K = k0+[16..31].  One contiguous 16-half (32B) load.
        const v16h a = *(const v16h*)(xrow + k0 + grp * 16);
        // B fragment (32x16): lane col = m; grp0 K=k0+[0..15], grp1 K=k0+[16..31]
        // -> 16 contiguous halfs (32B) in the column-major packed weights.
        const int kb = k0 + grp * 16;
        const v16h ba = *(const v16h*)(wacol + kb);
        const v16h bb = *(const v16h*)(wbcol + kb);
        ca = __builtin_amdgcn_wmma_f32_16x16x32_f16(false, a, false, ba, (short)0, ca, false, false);
        cb = __builtin_amdgcn_wmma_f32_16x16x32_f16(false, a, false, bb, (short)0, cb, false, false);
    }

    // C/D layout: lane col = m; VGPR r -> row r + grp*8
#pragma unroll
    for (int r = 0; r < 8; ++r) {
        const long o = (long)(rowTile * 16 + r + grp * 8) * C + col;
        Ya[o] = ca[r];
        Yb[o] = cb[r];
    }
}

// ---------------------------------------------------------------------------
// Layer-1 edge softmax numerators + denominators. Wave per edge.
// Lane handles channels [4L,4L+3]; head = L/4 (16 ch / head = 4 lanes).
// ---------------------------------------------------------------------------
__global__ __launch_bounds__(256)
void edge_softmax_l1(const float* __restrict__ XL, const float* __restrict__ XR,
                     const int* __restrict__ src, const int* __restrict__ dst,
                     const float* __restrict__ att,
                     float* __restrict__ ex, float* __restrict__ denom,
                     int E, int ET) {
    const int lane = threadIdx.x & 31;
    const int w = blockIdx.x * 8 + (threadIdx.x >> 5);
    if (w >= ET) return;
    int s, d;
    if (w < E) { s = src[w]; d = dst[w]; } else { s = d = w - E; }

    const float4 xl = ((const float4*)(XL + (long)s * 128))[lane];
    const float4 xr = ((const float4*)(XR + (long)d * 128))[lane];
    const float4 av = ((const float4*)att)[lane];
    float p = av.x * lrelu(xl.x + xr.x) + av.y * lrelu(xl.y + xr.y)
            + av.z * lrelu(xl.z + xr.z) + av.w * lrelu(xl.w + xr.w);
    p += __shfl_xor(p, 1, 32);
    p += __shfl_xor(p, 2, 32);
    if ((lane & 3) == 0) {
        const int h = lane >> 2;
        const float e = __expf(p);
        ex[(long)w * 8 + h] = e;
        atomAdd(&denom[(long)d * 8 + h], e);
    }
}

__global__ __launch_bounds__(256)
void edge_agg_l1(const float* __restrict__ XL,
                 const int* __restrict__ src, const int* __restrict__ dst,
                 const float* __restrict__ ex, const float* __restrict__ denom,
                 float* __restrict__ agg, int E, int ET) {
    const int lane = threadIdx.x & 31;
    const int w = blockIdx.x * 8 + (threadIdx.x >> 5);
    if (w >= ET) return;
    int s, d;
    if (w < E) { s = src[w]; d = dst[w]; } else { s = d = w - E; }
    const int h = lane >> 2;
    const float alpha = ex[(long)w * 8 + h] / denom[(long)d * 8 + h];
    const float4 xl = ((const float4*)(XL + (long)s * 128))[lane];
    float* o = agg + (long)d * 128 + lane * 4;
    atomAdd(o + 0, xl.x * alpha);
    atomAdd(o + 1, xl.y * alpha);
    atomAdd(o + 2, xl.z * alpha);
    atomAdd(o + 3, xl.w * alpha);
}

// agg + b1 -> ELU -> f16 row-major (feeds layer-2 WMMA). 4 elems/thread.
__global__ __launch_bounds__(256)
void bias_elu_f16(const float* __restrict__ a, const float* __restrict__ b,
                  _Float16* __restrict__ out, int total4) {
    const int i = blockIdx.x * 256 + threadIdx.x;
    if (i >= total4) return;
    const float4 v = ((const float4*)a)[i];
    const float4 bv = ((const float4*)b)[i & 31];      // row stride 128 = 32 float4
    float e0 = v.x + bv.x, e1 = v.y + bv.y, e2 = v.z + bv.z, e3 = v.w + bv.w;
    e0 = e0 > 0.f ? e0 : expm1f(e0);
    e1 = e1 > 0.f ? e1 : expm1f(e1);
    e2 = e2 > 0.f ? e2 : expm1f(e2);
    e3 = e3 > 0.f ? e3 : expm1f(e3);
    v4h o; o[0] = (_Float16)e0; o[1] = (_Float16)e1; o[2] = (_Float16)e2; o[3] = (_Float16)e3;
    ((v4h*)out)[i] = o;
}

// ---------------------------------------------------------------------------
// Layer-2 (heads=1, C=64). Wave per edge, 2 channels per lane.
// ---------------------------------------------------------------------------
__global__ __launch_bounds__(256)
void edge_softmax_l2(const float* __restrict__ XL, const float* __restrict__ XR,
                     const int* __restrict__ src, const int* __restrict__ dst,
                     const float* __restrict__ att,
                     float* __restrict__ ex, float* __restrict__ denom,
                     int E, int ET) {
    const int lane = threadIdx.x & 31;
    const int w = blockIdx.x * 8 + (threadIdx.x >> 5);
    if (w >= ET) return;
    int s, d;
    if (w < E) { s = src[w]; d = dst[w]; } else { s = d = w - E; }
    const float2 xl = ((const float2*)(XL + (long)s * 64))[lane];
    const float2 xr = ((const float2*)(XR + (long)d * 64))[lane];
    const float2 av = ((const float2*)att)[lane];
    float p = av.x * lrelu(xl.x + xr.x) + av.y * lrelu(xl.y + xr.y);
#pragma unroll
    for (int o = 1; o < 32; o <<= 1) p += __shfl_xor(p, o, 32);
    if (lane == 0) {
        const float e = __expf(p);
        ex[w] = e;
        atomAdd(&denom[d], e);
    }
}

__global__ __launch_bounds__(256)
void edge_agg_l2(const float* __restrict__ XL,
                 const int* __restrict__ src, const int* __restrict__ dst,
                 const float* __restrict__ ex, const float* __restrict__ denom,
                 float* __restrict__ out, int E, int ET) {
    const int lane = threadIdx.x & 31;
    const int w = blockIdx.x * 8 + (threadIdx.x >> 5);
    if (w >= ET) return;
    int s, d;
    if (w < E) { s = src[w]; d = dst[w]; } else { s = d = w - E; }
    const float alpha = ex[w] / denom[d];
    const float2 xl = ((const float2*)(XL + (long)s * 64))[lane];
    float* o = out + (long)d * 64 + lane * 2;
    atomAdd(o + 0, xl.x * alpha);
    atomAdd(o + 1, xl.y * alpha);
}

__global__ __launch_bounds__(256)
void log_softmax64(float* __restrict__ out, const float* __restrict__ b2, int N) {
    const int lane = threadIdx.x & 31;
    const int row = blockIdx.x * 8 + (threadIdx.x >> 5);
    if (row >= N) return;
    float2 v = ((const float2*)(out + (long)row * 64))[lane];
    v.x += b2[2 * lane];
    v.y += b2[2 * lane + 1];
    float m = fmaxf(v.x, v.y);
#pragma unroll
    for (int o = 1; o < 32; o <<= 1) m = fmaxf(m, __shfl_xor(m, o, 32));
    float s = __expf(v.x - m) + __expf(v.y - m);
#pragma unroll
    for (int o = 1; o < 32; o <<= 1) s += __shfl_xor(s, o, 32);
    const float lse = m + __logf(s);
    float2 r; r.x = v.x - lse; r.y = v.y - lse;
    ((float2*)(out + (long)row * 64))[lane] = r;
}

// ---------------------------------------------------------------------------
extern "C" void kernel_launch(void* const* d_in, const int* in_sizes, int n_in,
                              void* d_out, int out_size, void* d_ws, size_t ws_size,
                              hipStream_t stream) {
    (void)n_in; (void)out_size; (void)ws_size;
    const float* x    = (const float*)d_in[0];
    const int*   ei   = (const int*)  d_in[1];
    const float* Wl1  = (const float*)d_in[2];
    const float* Wr1  = (const float*)d_in[3];
    const float* att1 = (const float*)d_in[4];
    const float* b1   = (const float*)d_in[5];
    const float* Wl2  = (const float*)d_in[6];
    const float* Wr2  = (const float*)d_in[7];
    const float* att2 = (const float*)d_in[8];
    const float* b2   = (const float*)d_in[9];

    const int N  = in_sizes[0] / 128;
    const int E  = in_sizes[1] / 2;
    const int ET = E + N;                       // with self-loops
    const int* srcA = ei;
    const int* dstA = ei + E;

    char* ws = (char*)d_ws;
    size_t off = 0;
    auto alloc = [&](size_t bytes) -> void* {
        char* p = ws + off;
        off = (off + bytes + 255) & ~(size_t)255;
        return p;
    };
    _Float16* x16   = (_Float16*)alloc((size_t)N * 128 * 2);
    _Float16* h16   = (_Float16*)alloc((size_t)N * 128 * 2);
    _Float16* Wl1T  = (_Float16*)alloc((size_t)128 * 128 * 2);
    _Float16* Wr1T  = (_Float16*)alloc((size_t)128 * 128 * 2);
    _Float16* Wl2T  = (_Float16*)alloc((size_t)128 * 64 * 2);
    _Float16* Wr2T  = (_Float16*)alloc((size_t)128 * 64 * 2);
    float* XL1  = (float*)alloc((size_t)N * 128 * 4);
    float* XR1  = (float*)alloc((size_t)N * 128 * 4);
    float* ex1  = (float*)alloc((size_t)ET * 8 * 4);
    float* dn1  = (float*)alloc((size_t)N * 8 * 4);
    float* agg1 = (float*)alloc((size_t)N * 128 * 4);
    float* XL2  = (float*)alloc((size_t)N * 64 * 4);
    float* XR2  = (float*)alloc((size_t)N * 64 * 4);
    float* ex2  = (float*)alloc((size_t)ET * 4);
    float* dn2  = (float*)alloc((size_t)N * 4);
    float* out  = (float*)d_out;

    hipMemsetAsync(dn1,  0, (size_t)N * 8 * 4,   stream);
    hipMemsetAsync(agg1, 0, (size_t)N * 128 * 4, stream);
    hipMemsetAsync(dn2,  0, (size_t)N * 4,       stream);
    hipMemsetAsync(out,  0, (size_t)N * 64 * 4,  stream);

    // ---- pack operands to f16 (x row-major; weights column-major) ----
    const int nx4 = N * 128 / 4;
    f32_to_f16_4<<<(nx4 + 255) / 256, 256, 0, stream>>>(x, x16, nx4);
    pack_w_t<<<(128 * 128 + 255) / 256, 256, 0, stream>>>(Wl1, Wl1T, 128, 128);
    pack_w_t<<<(128 * 128 + 255) / 256, 256, 0, stream>>>(Wr1, Wr1T, 128, 128);
    pack_w_t<<<(128 * 64 + 255) / 256, 256, 0, stream>>>(Wl2, Wl2T, 128, 64);
    pack_w_t<<<(128 * 64 + 255) / 256, 256, 0, stream>>>(Wr2, Wr2T, 128, 64);

    const int rt = (N + 15) / 16;               // 3125 row tiles
    const dim3 blk(128);
    const int eb = (ET + 7) / 8;                // wave-per-edge blocks

    // Layer 1
    gemm_dual_wmma<<<dim3((rt + 3) / 4, 8), blk, 0, stream>>>(x16, Wl1T, Wr1T, XL1, XR1, rt, 128);
    edge_softmax_l1<<<eb, 256, 0, stream>>>(XL1, XR1, srcA, dstA, att1, ex1, dn1, E, ET);
    edge_agg_l1<<<eb, 256, 0, stream>>>(XL1, srcA, dstA, ex1, dn1, agg1, E, ET);
    bias_elu_f16<<<(nx4 + 255) / 256, 256, 0, stream>>>(agg1, b1, h16, nx4);

    // Layer 2
    gemm_dual_wmma<<<dim3((rt + 3) / 4, 4), blk, 0, stream>>>(h16, Wl2T, Wr2T, XL2, XR2, rt, 64);
    edge_softmax_l2<<<eb, 256, 0, stream>>>(XL2, XR2, srcA, dstA, att2, ex2, dn2, E, ET);
    edge_agg_l2<<<eb, 256, 0, stream>>>(XL2, srcA, dstA, ex2, dn2, out, E, ET);
    log_softmax64<<<(N + 7) / 8, 256, 0, stream>>>(out, b2, N);
}
